// StackedAttentionLSTM_80350248174037
// MI455X (gfx1250) — matrix operand order repaired and checked by reading, not compiled, not run on hardware
//
#include <hip/hip_runtime.h>

// StackedAttentionLSTM for MI455X (gfx1250): bf16 WMMA GEMMs + VALU attention.
// B=32, T=256, S=512, H=512, L=2.

#define B_  32
#define T_  256
#define S_  512
#define H_  512
#define L_  2
#define G4  (4 * H_)  // 2048

typedef __attribute__((ext_vector_type(16))) __bf16 v16bf;
typedef __attribute__((ext_vector_type(8)))  float  v8f;

__device__ __forceinline__ unsigned short f2bf(float f) {
    union { float f; unsigned u; } v; v.f = f;
    unsigned r = 0x7FFFu + ((v.u >> 16) & 1u);
    return (unsigned short)((v.u + r) >> 16);
}

// Inverse of the 16-bit A-matrix lane/element mapping (ISA 7.12.2):
// given k%32, produce lane-half (hi) and element index j within the v16bf.
__device__ __forceinline__ void a_inv(int kl, int& hi, int& j) {
    if (kl < 8)       { hi = 0; j = kl; }
    else if (kl < 16) { hi = 1; j = kl - 8; }
    else if (kl < 24) { hi = 0; j = kl - 8; }
    else              { hi = 1; j = kl - 16; }
}

// Pack fp32 activations (rows x K, row stride given) into WMMA A-tile bf16
// layout: dst[((mt*KtTot + ktStart+kt)*32 + lane)*16 + j].
__global__ void pack_a_kernel(const float* __restrict__ src,
                              unsigned short* __restrict__ dst,
                              int Mtiles, int KtTot, int ktStart, int ktCnt,
                              int rowStride) {
    int idx = blockIdx.x * blockDim.x + threadIdx.x;
    int total = Mtiles * ktCnt * 512;
    if (idx >= total) return;
    int j = idx & 15, lane = (idx >> 4) & 31, rem = idx >> 9;
    int ktr = rem % ktCnt, mt = rem / ktCnt;
    int m  = mt * 16 + (lane & 15);
    int hi = lane >> 4;
    int k  = ktr * 32 + ((j < 8) ? j : j + 8) + hi * 8;
    dst[(((size_t)(mt * KtTot) + ktStart + ktr) * 32 + lane) * 16 + j] =
        f2bf(src[(size_t)m * rowStride + k]);
}

// Pack fp32 weights (K x N, row-major, leading dim ldn) into WMMA B-tile bf16
// layout: lanes 0-15 = cols, K 0..7,16..23; lanes 16-31 same cols, K 8..15,24..31.
__global__ void pack_b_kernel(const float* __restrict__ src,
                              unsigned short* __restrict__ dst,
                              int Ntiles, int KtTot, int ktStart, int ktCnt,
                              int ldn) {
    int idx = blockIdx.x * blockDim.x + threadIdx.x;
    int total = Ntiles * ktCnt * 512;
    if (idx >= total) return;
    int j = idx & 15, lane = (idx >> 4) & 31, rem = idx >> 9;
    int ktr = rem % ktCnt, nt = rem / ktCnt;
    int n = nt * 16 + (lane & 15);
    int k = ktr * 32 + ((j < 8) ? j : j + 8) + ((lane >> 4) << 3);
    dst[(((size_t)(nt * KtTot) + ktStart + ktr) * 32 + lane) * 16 + j] =
        f2bf(src[(size_t)k * ldn + n]);
}

// Generic bf16 WMMA GEMM over pre-packed tiles: one wave per 16x16 C tile.
// C[row, col] = sum_k A[row,k]*B[k,col] + bias0[col] + bias1[col].
__global__ __launch_bounds__(256) void wmma_gemm_kernel(
        const unsigned short* __restrict__ Apk,
        const unsigned short* __restrict__ Bpk,
        const float* __restrict__ bias0, const float* __restrict__ bias1,
        float* __restrict__ C,
        int Mtiles, int Ntiles,
        int KtA, int a0, int KtB, int b0, int cnt, int ldc) {
    int tid  = blockIdx.x * blockDim.x + threadIdx.x;
    int wave = tid >> 5, lane = tid & 31;
    if (wave >= Mtiles * Ntiles) return;
    int mt = wave / Ntiles, nt = wave % Ntiles;
    v8f acc = {};
    const unsigned short* ap = Apk + ((size_t)(mt * KtA + a0) * 32 + lane) * 16;
    const unsigned short* bp = Bpk + ((size_t)(nt * KtB + b0) * 32 + lane) * 16;
    for (int kt = 0; kt < cnt; ++kt) {
        v16bf a = *reinterpret_cast<const v16bf*>(ap);
        v16bf b = *reinterpret_cast<const v16bf*>(bp);
        if (kt + 1 < cnt) {
            __builtin_prefetch(ap + 512, 0, 3);  // WGP-scope prefetch of next K-tile
            __builtin_prefetch(bp + 512, 0, 3);
        }
        acc = __builtin_amdgcn_wmma_f32_16x16x32_bf16(
            false, a, false, b, (short)0, acc, false, false);
        ap += 512; bp += 512;
    }
    int n  = nt * 16 + (lane & 15);
    int mb = (lane >> 4) * 8;
    float badd = 0.f;
    if (bias0) badd += bias0[n];
    if (bias1) badd += bias1[n];
#pragma unroll
    for (int r = 0; r < 8; ++r) {
        int row = mt * 16 + mb + r;
        C[(size_t)row * ldc + n] = acc[r] + badd;
    }
}

// LSTM pointwise: gates (B x 4H) -> cy (in-place c_buf), hy packed straight
// into the [wc, hy] A-buffer (K-tiles 16..31).
__global__ void lstm_pointwise_kernel(const float* __restrict__ gates,
                                      float* __restrict__ c_buf,
                                      unsigned short* __restrict__ concat2) {
    int idx = blockIdx.x * blockDim.x + threadIdx.x;
    if (idx >= B_ * H_) return;
    int b = idx >> 9, h = idx & 511;
    const float* g = gates + (size_t)b * G4;
    float ig = 1.f / (1.f + __expf(-g[h]));
    float fg = 1.f / (1.f + __expf(-g[512 + h]));
    float gg = tanhf(g[1024 + h]);
    float og = 1.f / (1.f + __expf(-g[1536 + h]));
    float cy = fg * c_buf[idx] + ig * gg;
    c_buf[idx] = cy;
    float hy = og * tanhf(cy);
    int hi, j; a_inv(h & 31, hi, j);
    int off = ((((b >> 4) * 32) + 16 + (h >> 5)) * 32 + ((b & 15) | (hi << 4))) * 16 + j;
    concat2[off] = f2bf(hy);
}

// Per-batch attention: scores = ctx_b . target, softmax, wc = attn . ctx_b.
// wc packed into [wc, hy] A-buffer (K-tiles 0..15). att written transposed.
__global__ __launch_bounds__(256) void attn_kernel(
        const float* __restrict__ ctx, const float* __restrict__ target,
        unsigned short* __restrict__ concat2,
        float* __restrict__ att_out, int t) {
    __shared__ float tgt[512];
    __shared__ float sc[512];
    __shared__ float red[256];
    __shared__ float wcs[512];
    int b = blockIdx.x, tid = threadIdx.x;
    tgt[tid]       = target[(size_t)b * 512 + tid];
    tgt[tid + 256] = target[(size_t)b * 512 + tid + 256];
    __syncthreads();
    int lane = tid & 31, wave = tid >> 5;
    for (int s = wave; s < 512; s += 8) {
        const float* cr = ctx + ((size_t)s * B_ + b) * H_;
        float acc = 0.f;
        for (int h = lane; h < 512; h += 32) acc += cr[h] * tgt[h];
        for (int m = 16; m >= 1; m >>= 1) acc += __shfl_xor(acc, m, 32);
        if (lane == 0) sc[s] = acc;
    }
    __syncthreads();
    float mx = fmaxf(sc[tid], sc[tid + 256]);
    red[tid] = mx; __syncthreads();
    for (int st = 128; st >= 1; st >>= 1) {
        if (tid < st) red[tid] = fmaxf(red[tid], red[tid + st]);
        __syncthreads();
    }
    mx = red[0]; __syncthreads();
    float e0 = __expf(sc[tid] - mx), e1 = __expf(sc[tid + 256] - mx);
    red[tid] = e0 + e1; __syncthreads();
    for (int st = 128; st >= 1; st >>= 1) {
        if (tid < st) red[tid] += red[tid + st];
        __syncthreads();
    }
    float inv = 1.f / red[0]; __syncthreads();
    sc[tid] = e0 * inv; sc[tid + 256] = e1 * inv;
    __syncthreads();
    if (att_out) {  // att[s, t*B + b]
        att_out[(size_t)tid * (T_ * B_) + t * B_ + b]         = sc[tid];
        att_out[(size_t)(tid + 256) * (T_ * B_) + t * B_ + b] = sc[tid + 256];
    }
    float w0 = 0.f, w1 = 0.f;
    for (int s = 0; s < 512; ++s) {
        const float* cr = ctx + ((size_t)s * B_ + b) * H_;
        float a = sc[s];
        w0 += a * cr[tid];
        w1 += a * cr[tid + 256];
    }
    wcs[tid] = w0; wcs[tid + 256] = w1;
    __syncthreads();
    for (int p = tid; p < 512; p += 256) {
        int kt = p >> 5, rest = p & 31, hi = rest >> 4, j = rest & 15;
        int k  = kt * 32 + ((j < 8) ? j : j + 8) + hi * 8;
        int off = ((((b >> 4) * 32) + kt) * 32 + ((b & 15) | (hi << 4))) * 16 + j;
        concat2[off] = f2bf(wcs[k]);
    }
}

// Output GEMM: h~ = tanh([wc,hy] @ Wa_out). Writes fp32 activation (layer
// output) and packs h~ as the h-half of next step's [x,h] A-buffer.
__global__ __launch_bounds__(256) void gemm_tanh_out_kernel(
        const unsigned short* __restrict__ Apk,   // concat2, KtA=32
        const unsigned short* __restrict__ Bpk,   // Wa_out packed, KtB=32
        float* __restrict__ xdst, int strideB,
        unsigned short* __restrict__ concat1_h) {
    int tid  = blockIdx.x * blockDim.x + threadIdx.x;
    int wave = tid >> 5, lane = tid & 31;
    if (wave >= 2 * 32) return;
    int mt = wave >> 5, nt = wave & 31;
    v8f acc = {};
    const unsigned short* ap = Apk + ((size_t)(mt * 32) * 32 + lane) * 16;
    const unsigned short* bp = Bpk + ((size_t)(nt * 32) * 32 + lane) * 16;
    for (int kt = 0; kt < 32; ++kt) {
        v16bf a  = *reinterpret_cast<const v16bf*>(ap);
        v16bf bb = *reinterpret_cast<const v16bf*>(bp);
        if (kt + 1 < 32) {
            __builtin_prefetch(ap + 512, 0, 3);
            __builtin_prefetch(bp + 512, 0, 3);
        }
        acc = __builtin_amdgcn_wmma_f32_16x16x32_bf16(
            false, a, false, bb, (short)0, acc, false, false);
        ap += 512; bp += 512;
    }
    int n  = nt * 16 + (lane & 15);
    int mb = (lane >> 4) * 8;
#pragma unroll
    for (int r = 0; r < 8; ++r) {
        int b = mt * 16 + mb + r;
        float h = tanhf(acc[r]);
        xdst[(size_t)b * strideB + n] = h;
        int hi, j; a_inv(n & 31, hi, j);
        int off = ((((b >> 4) * 32) + 16 + (n >> 5)) * 32 + ((b & 15) | (hi << 4))) * 16 + j;
        concat1_h[off] = f2bf(h);
    }
}

__global__ void copy_kernel(const float* __restrict__ src, float* __restrict__ dst, int n) {
    int i = blockIdx.x * blockDim.x + threadIdx.x;
    if (i < n) dst[i] = src[i];
}

__global__ void finalize_kernel(const float* __restrict__ hsrc, int strideB,
                                const float* __restrict__ c_buf,
                                float* __restrict__ out_h, float* __restrict__ out_c) {
    int idx = blockIdx.x * blockDim.x + threadIdx.x;
    if (idx >= B_ * H_) return;
    int b = idx >> 9, h = idx & 511;
    out_h[idx] = hsrc[(size_t)b * strideB + h];
    out_c[idx] = c_buf[idx];
}

extern "C" void kernel_launch(void* const* d_in, const int* in_sizes, int n_in,
                              void* d_out, int out_size, void* d_ws, size_t ws_size,
                              hipStream_t stream) {
    (void)in_sizes; (void)n_in; (void)out_size; (void)ws_size;
    const float* input  = (const float*)d_in[0];   // (B,T,H)
    const float* h0     = (const float*)d_in[1];   // (B,H)
    const float* c0     = (const float*)d_in[2];   // (B,H)
    const float* ctx    = (const float*)d_in[3];   // (S,B,H)
    const float* Wi     = (const float*)d_in[4];   // (L,H,4H)
    const float* bi     = (const float*)d_in[5];   // (L,4H)
    const float* Wh     = (const float*)d_in[6];   // (L,H,4H)
    const float* bh     = (const float*)d_in[7];   // (L,4H)
    const float* Wa_in  = (const float*)d_in[8];   // (L,H,H)
    const float* Wa_out = (const float*)d_in[9];   // (L,2H,H)

    float* out     = (float*)d_out;
    float* out_x   = out;                                  // (B,T,H)
    float* out_h   = out   + (size_t)B_ * T_ * H_;         // (L,B,H)
    float* out_c   = out_h + (size_t)L_ * B_ * H_;         // (L,B,H)
    float* out_att = out_c + (size_t)L_ * B_ * H_;         // (S, T*B)

    char* ws = (char*)d_ws;
    size_t off = 0;
    auto alloc = [&](size_t bytes) -> void* {
        void* p = ws + off;
        off = (off + bytes + 255) & ~(size_t)255;
        return p;
    };
    unsigned short* concat1 = (unsigned short*)alloc((size_t)B_ * 1024 * 2); // [x_t, h]
    unsigned short* concat2 = (unsigned short*)alloc((size_t)B_ * 1024 * 2); // [wc, hy]
    unsigned short* WB1 = (unsigned short*)alloc((size_t)1024 * 2048 * 2);   // [Wi;Wh]
    unsigned short* WB2 = (unsigned short*)alloc((size_t)512 * 512 * 2);     // Wa_in
    unsigned short* WB3 = (unsigned short*)alloc((size_t)1024 * 512 * 2);    // Wa_out
    float* gates  = (float*)alloc((size_t)B_ * G4 * 4);
    float* c_buf  = (float*)alloc((size_t)B_ * H_ * 4);
    float* target = (float*)alloc((size_t)B_ * H_ * 4);
    float* X_buf  = (float*)alloc((size_t)T_ * B_ * H_ * 4);  // layer-0 outputs

    dim3 blk(256);
    for (int l = 0; l < L_; ++l) {
        const float* Wi_l     = Wi + (size_t)l * H_ * G4;
        const float* Wh_l     = Wh + (size_t)l * H_ * G4;
        const float* bi_l     = bi + (size_t)l * G4;
        const float* bh_l     = bh + (size_t)l * G4;
        const float* Wa_in_l  = Wa_in  + (size_t)l * H_ * H_;
        const float* Wa_out_l = Wa_out + (size_t)l * 2 * H_ * H_;

        // Per-layer weight repack into WMMA B-tile bf16 layouts.
        pack_b_kernel<<<(128 * 16 * 512) / 256, blk, 0, stream>>>(Wi_l, WB1, 128, 32, 0, 16, G4);
        pack_b_kernel<<<(128 * 16 * 512) / 256, blk, 0, stream>>>(Wh_l, WB1, 128, 32, 16, 16, G4);
        pack_b_kernel<<<(32 * 16 * 512) / 256, blk, 0, stream>>>(Wa_in_l, WB2, 32, 16, 0, 16, H_);
        pack_b_kernel<<<(32 * 32 * 512) / 256, blk, 0, stream>>>(Wa_out_l, WB3, 32, 32, 0, 32, H_);
        // State init: h0 into the h-half of [x,h]; c0 into c_buf.
        pack_a_kernel<<<(2 * 16 * 512) / 256, blk, 0, stream>>>(h0, concat1, 2, 32, 16, 16, H_);
        copy_kernel<<<(B_ * H_) / 256, blk, 0, stream>>>(c0, c_buf, B_ * H_);

        const float* xsrc = (l == 0) ? input : X_buf;
        int x_sB   = (l == 0) ? T_ * H_ : H_;       // row(b) stride in source
        int x_tOff = (l == 0) ? H_ : B_ * H_;       // t stride in source
        float* xdst = (l == L_ - 1) ? out_x : X_buf;
        int d_sB    = (l == L_ - 1) ? T_ * H_ : H_;
        int d_tOff  = (l == L_ - 1) ? H_ : B_ * H_;
        float* att  = (l == L_ - 1) ? out_att : nullptr;

        for (int t = 0; t < T_; ++t) {
            // x_t into the x-half of [x,h]
            pack_a_kernel<<<64, blk, 0, stream>>>(xsrc + (size_t)t * x_tOff,
                                                  concat1, 2, 32, 0, 16, x_sB);
            // gates = [x,h] @ [Wi;Wh] + bi + bh   (M=32, K=1024, N=2048)
            wmma_gemm_kernel<<<32, blk, 0, stream>>>(concat1, WB1, bi_l, bh_l, gates,
                                                     2, 128, 32, 0, 32, 0, 32, G4);
            lstm_pointwise_kernel<<<64, blk, 0, stream>>>(gates, c_buf, concat2);
            // target = hy @ Wa_in   (A = hy half of concat2)
            wmma_gemm_kernel<<<8, blk, 0, stream>>>(concat2, WB2, nullptr, nullptr, target,
                                                    2, 32, 32, 16, 16, 0, 16, H_);
            attn_kernel<<<32, blk, 0, stream>>>(ctx, target, concat2, att, t);
            // h~ = tanh([wc,hy] @ Wa_out) -> layer output + packed h for next step
            gemm_tanh_out_kernel<<<8, blk, 0, stream>>>(concat2, WB3,
                                                        xdst + (size_t)t * d_tOff, d_sB,
                                                        concat1);
        }
        finalize_kernel<<<64, blk, 0, stream>>>(xdst + (size_t)(T_ - 1) * d_tOff, d_sB,
                                                c_buf,
                                                out_h + (size_t)l * B_ * H_,
                                                out_c + (size_t)l * B_ * H_);
    }
}
